// RelativePositionEncoderLayer_38714835206832
// MI455X (gfx1250) — compile-verified, matrix operand order checked by reading, not run on hardware
//
#include <hip/hip_runtime.h>
#include <hip/hip_bf16.h>
#include <math.h>

#define BB 2
#define SS 2048
#define DD 1024
#define HH 16
#define DHH 64
#define FFD 4096
#define TOK (BB * SS)

typedef __attribute__((ext_vector_type(16))) __bf16 v16bf;
typedef __attribute__((ext_vector_type(8)))  float  v8f;
typedef unsigned u32x4 __attribute__((ext_vector_type(4)));
typedef unsigned u32x8 __attribute__((ext_vector_type(8)));

// ---------------------------------------------------------------------------
// WMMA helpers (CDNA5: V_WMMA_F32_16X16X32_BF16, wave32)
// ---------------------------------------------------------------------------
__device__ __forceinline__ v8f wmma_bf16(v16bf a, v16bf b, v8f c) {
  return __builtin_amdgcn_wmma_f32_16x16x32_bf16(false, a, false, b, (short)0, c,
                                                 false, false);
}

// A-operand layout (16x32 bf16, MxK): lane L holds row M=L&15; element e holds
// K = (e/8)*16 + (L>>4)*8 + (e&7).  Two contiguous 16B chunks per lane.
// B-operand (32x16, KxN) uses the identical packing with N in place of M, i.e.
// it is read from a row-major [N][K] ("transposed") matrix.
__device__ __forceinline__ v16bf ld_frag(const __bf16* base, int ld, int r0, int k0) {
  int lane = threadIdx.x & 31;
  int row  = r0 + (lane & 15);
  int kb   = k0 + ((lane >> 4) << 3);
  const __bf16* p = base + (size_t)row * ld + kb;
  union { float4 q[2]; v16bf v; } u;
  u.q[0] = *(const float4*)(p);
  u.q[1] = *(const float4*)(p + 16);
  return u.v;
}

__device__ __forceinline__ v16bf ld_frag_clamp(const __bf16* base, int ld, int r0,
                                               int k0, int rmax) {
  int lane = threadIdx.x & 31;
  int row  = r0 + (lane & 15);
  row = row < 0 ? 0 : (row > rmax ? rmax : row);
  int kb = k0 + ((lane >> 4) << 3);
  const __bf16* p = base + (size_t)row * ld + kb;
  union { float4 q[2]; v16bf v; } u;
  u.q[0] = *(const float4*)(p);
  u.q[1] = *(const float4*)(p + 16);
  return u.v;
}

// ---------------------------------------------------------------------------
// Tensor Data Mover: DMA a 64x64 bf16 tile (row stride = stride_elems) from
// global memory into LDS (packed row-major, 64*64*2 bytes).  One instruction,
// tracked by TENSORcnt.  All descriptor inputs are wave-uniform -> SGPRs.
// D# group0: count=1 | lds_addr | global_addr | type=2
// D# group1: data_size=2B, tensor_dim0=64, tensor_dim1=64, tile 64x64,
//            tensor_dim0_stride=stride_elems
// ---------------------------------------------------------------------------
__device__ __forceinline__ void tdm_load_64x64_bf16(unsigned lds_off,
                                                    const __bf16* gptr,
                                                    unsigned stride_elems) {
  unsigned long long ga = (unsigned long long)gptr;
  u32x4 g0;
  g0[0] = 1u;                                            // count = 1 (valid)
  g0[1] = lds_off;                                       // lds_addr (bytes)
  g0[2] = (unsigned)(ga & 0xFFFFFFFFu);                  // global_addr[31:0]
  g0[3] = (unsigned)((ga >> 32) & 0x01FFFFFFu) | (2u << 30);  // addr hi | type=2
  u32x8 g1;
  g1[0] = 1u << 16;        // workgroup_mask=0, data_size=1 (2 bytes)
  g1[1] = 64u << 16;       // tensor_dim0[15:0] = 64  (bits 79:48)
  g1[2] = 64u << 16;       // tensor_dim0 hi = 0 | tensor_dim1[15:0] = 64
  g1[3] = 64u << 16;       // tensor_dim1 hi = 0 | tile_dim0 = 64
  g1[4] = 64u;             // tile_dim1 = 64, tile_dim2 = 0
  g1[5] = stride_elems;    // tensor_dim0_stride low 32 (data_size units)
  g1[6] = 0u;              // stride hi | tensor_dim1_stride lo
  g1[7] = 0u;
  asm volatile("tensor_load_to_lds %0, %1" ::"s"(g0), "s"(g1) : "memory");
}

// ---------------------------------------------------------------------------
// fp32 -> bf16 weight transpose:  Wt[n*K + k] = W[k*N + n]
// ---------------------------------------------------------------------------
__global__ __launch_bounds__(256) void transpose_cvt(const float* __restrict__ W,
                                                     __bf16* __restrict__ Wt,
                                                     int K, int N) {
  int idx = blockIdx.x * 256 + threadIdx.x;
  if (idx >= K * N) return;
  int k = idx / N, n = idx - k * N;
  Wt[(size_t)n * K + k] = (__bf16)W[idx];
}

// Es[r][d] = Er[S-1-r][d]  (reversed relative-position table, bf16)
__global__ __launch_bounds__(256) void build_es(const float* __restrict__ Er,
                                                __bf16* __restrict__ Es) {
  int idx = blockIdx.x * 256 + threadIdx.x;  // r*64 + d
  int r = idx >> 6, d = idx & 63;
  Es[idx] = (__bf16)Er[(size_t)(SS - 1 - r) * DHH + d];
}

// ---------------------------------------------------------------------------
// LayerNorm over D=1024; one block per row; writes fp32 + bf16 copies
// ---------------------------------------------------------------------------
__global__ __launch_bounds__(256) void layernorm_kernel(
    const float* __restrict__ x, const float* __restrict__ g,
    const float* __restrict__ bta, float* __restrict__ of,
    __bf16* __restrict__ ob) {
  int row = blockIdx.x;
  int t   = threadIdx.x;
  const float* xr = x + (size_t)row * DD;
  float4 xv = ((const float4*)xr)[t];
  float s  = xv.x + xv.y + xv.z + xv.w;
  float ss = xv.x * xv.x + xv.y * xv.y + xv.z * xv.z + xv.w * xv.w;
#pragma unroll
  for (int off = 1; off < 32; off <<= 1) {
    s  += __shfl_xor(s, off, 32);
    ss += __shfl_xor(ss, off, 32);
  }
  __shared__ float rs[8], rss[8];
  int w = t >> 5;
  if ((t & 31) == 0) { rs[w] = s; rss[w] = ss; }
  __syncthreads();
  if (t == 0) {
    float a = 0.f, b = 0.f;
    for (int i = 0; i < 8; ++i) { a += rs[i]; b += rss[i]; }
    rs[0] = a; rss[0] = b;
  }
  __syncthreads();
  float mu   = rs[0] * (1.f / DD);
  float var  = rss[0] * (1.f / DD) - mu * mu;
  float rstd = rsqrtf(var + 1e-5f);
  int c0 = t * 4;
  float4 yv;
  yv.x = (xv.x - mu) * rstd * g[c0 + 0] + bta[c0 + 0];
  yv.y = (xv.y - mu) * rstd * g[c0 + 1] + bta[c0 + 1];
  yv.z = (xv.z - mu) * rstd * g[c0 + 2] + bta[c0 + 2];
  yv.w = (xv.w - mu) * rstd * g[c0 + 3] + bta[c0 + 3];
  ((float4*)(of + (size_t)row * DD))[t] = yv;
  __bf16* obr = ob + (size_t)row * DD + c0;
  obr[0] = (__bf16)yv.x; obr[1] = (__bf16)yv.y;
  obr[2] = (__bf16)yv.z; obr[3] = (__bf16)yv.w;
}

// ---------------------------------------------------------------------------
// WMMA GEMM: C[M,N] = A[M,K](bf16) @ Bt[N,K](bf16)^T, fused epilogue.
// Block = 128 threads (4 waves, 2x2), 64x128 block tile, 32x64 per wave:
// 8 independent accumulator chains per wave (hides XDL latency), 12 b128
// loads per 8 WMMAs (~21 flops per L2 byte; all operands L2-resident).
// ---------------------------------------------------------------------------
__global__ __launch_bounds__(128) void gemm_bf16_kernel(
    const __bf16* __restrict__ A, const __bf16* __restrict__ Bt,
    const float* __restrict__ bias, const float* __restrict__ residual,
    float* __restrict__ outf, __bf16* __restrict__ outb,
    int M, int N, int K, float scale, int relu) {
  int w    = threadIdx.x >> 5;
  int lane = threadIdx.x & 31;
  int hi   = lane >> 4, Nl = lane & 15;
  int m0   = blockIdx.y * 64 + (w >> 1) * 32;
  int n0   = blockIdx.x * 128 + (w & 1) * 64;
  v8f acc[2][4] = {};
  for (int k0 = 0; k0 < K; k0 += 32) {
    v16bf a0 = ld_frag(A, K, m0, k0);
    v16bf a1 = ld_frag(A, K, m0 + 16, k0);
    v16bf b0 = ld_frag(Bt, K, n0, k0);
    v16bf b1 = ld_frag(Bt, K, n0 + 16, k0);
    v16bf b2 = ld_frag(Bt, K, n0 + 32, k0);
    v16bf b3 = ld_frag(Bt, K, n0 + 48, k0);
    acc[0][0] = wmma_bf16(a0, b0, acc[0][0]);
    acc[0][1] = wmma_bf16(a0, b1, acc[0][1]);
    acc[0][2] = wmma_bf16(a0, b2, acc[0][2]);
    acc[0][3] = wmma_bf16(a0, b3, acc[0][3]);
    acc[1][0] = wmma_bf16(a1, b0, acc[1][0]);
    acc[1][1] = wmma_bf16(a1, b1, acc[1][1]);
    acc[1][2] = wmma_bf16(a1, b2, acc[1][2]);
    acc[1][3] = wmma_bf16(a1, b3, acc[1][3]);
  }
#pragma unroll
  for (int tm = 0; tm < 2; ++tm)
#pragma unroll
    for (int tn = 0; tn < 4; ++tn)
#pragma unroll
      for (int r = 0; r < 8; ++r) {
        int row = m0 + tm * 16 + r + 8 * hi;     // WMMA C layout: M = reg + 8*hi
        int col = n0 + tn * 16 + Nl;             //               N = lane & 15
        float v = (acc[tm][tn][r] + bias[col]) * scale;
        if (relu) v = fmaxf(v, 0.f);
        size_t idx = (size_t)row * N + col;
        if (residual) v += residual[idx];
        if (outf) outf[idx] = v;
        if (outb) outb[idx] = (__bf16)v;
      }
}

// ---------------------------------------------------------------------------
// Flash attention with relative-position skew.
//   att[i,j] = q_i . k_j + (j<=i ? q_i . Es[i-j] : 0),  Es[r]=Er[S-1-r]
// Block = 128 threads (4 waves); wave w owns 16 query rows; 64 keys/iter.
// K tile staged into LDS by the Tensor Data Mover (one tensor_load_to_lds per
// iteration, TENSORcnt-tracked, overlapped with the V-transpose staging).
// Skew: two extra WMMA products per 16x16 tile over Es row windows
// [rm-16,rm) and [rm,rm+16); the diagonal element (r = rm + M - N) for the
// lane holding S(M,N) sits in lane hi*16 + ((M-N)&15), same reg ->
// one ds_bpermute_b32 per accumulator reg.
// ---------------------------------------------------------------------------
__global__ __launch_bounds__(128) void flash_attn_kernel(
    const __bf16* __restrict__ q, const __bf16* __restrict__ k,
    const __bf16* __restrict__ v, const __bf16* __restrict__ Es,
    __bf16* __restrict__ attnb) {
  __shared__ __bf16 Kl[64 * 64];       // K tile, row-major [j][dh] (TDM dest)
  __shared__ __bf16 Vt[64 * 72];       // V transposed: [dh][j], padded stride
  __shared__ __bf16 Pt[4][16 * 64];    // per-wave P~ staging (C->A relayout)

  int b  = blockIdx.z, h = blockIdx.y;
  int i0 = blockIdx.x * 64;
  int w    = threadIdx.x >> 5;
  int lane = threadIdx.x & 31;
  int hi   = lane >> 4, Nl = lane & 15;
  int iw0  = i0 + w * 16;

  const __bf16* qb = q + (size_t)b * SS * DD + h * DHH;
  const __bf16* kb = k + (size_t)b * SS * DD + h * DHH;
  const __bf16* vb = v + (size_t)b * SS * DD + h * DHH;
  unsigned kl_off = (unsigned)(size_t)(&Kl[0]);

  // Q fragments live in registers for the whole kernel (row stride = D)
  v16bf aq0 = ld_frag(qb, DD, iw0, 0);
  v16bf aq1 = ld_frag(qb, DD, iw0, 32);

  v8f  o[4] = {};
  float m8[8], l8[8];
#pragma unroll
  for (int r = 0; r < 8; ++r) { m8[r] = -1e30f; l8[r] = 0.f; }

  for (int j0 = 0; j0 < SS; j0 += 64) {
    // ---- TDM: async-stage K tile into LDS (wave 0 issues the descriptor) ----
    if (w == 0)
      tdm_load_64x64_bf16(kl_off, kb + (size_t)j0 * DD, DD);

    // ---- meanwhile: stage V tile transposed into LDS (all 128 threads) ----
#pragma unroll
    for (int rep = 0; rep < 4; ++rep) {
      int c = threadIdx.x + rep * 128;  // 0..511
      int j = c >> 3, d8 = (c & 7) * 8;
      float4 d4 = *(const float4*)(vb + (size_t)(j0 + j) * DD + d8);
      const __bf16* e = (const __bf16*)&d4;
#pragma unroll
      for (int t = 0; t < 8; ++t) Vt[(d8 + t) * 72 + j] = e[t];
    }
    if (j0 + 64 < SS)
      __builtin_prefetch(vb + (size_t)(j0 + 64) * DD, 0, 1);
    if (w == 0)
      __builtin_amdgcn_s_wait_tensorcnt(0);
    __syncthreads();

    // ---- S = Q K^T + skew for 4 key micro-tiles (K frags from LDS) ----
    v8f stile[4];
#pragma unroll
    for (int tj = 0; tj < 4; ++tj) {
      int jt = j0 + tj * 16;
      v16bf kf0 = ld_frag(&Kl[0], 64, tj * 16, 0);
      v16bf kf1 = ld_frag(&Kl[0], 64, tj * 16, 32);
      v8f s = {};
      s = wmma_bf16(aq0, kf0, s);
      s = wmma_bf16(aq1, kf1, s);

      int rm = iw0 - jt;  // diagonal base of this tile
      v16bf ea0 = ld_frag_clamp(Es, DHH, rm - 16, 0,  SS - 1);
      v16bf ea1 = ld_frag_clamp(Es, DHH, rm - 16, 32, SS - 1);
      v16bf eb0 = ld_frag_clamp(Es, DHH, rm,      0,  SS - 1);
      v16bf eb1 = ld_frag_clamp(Es, DHH, rm,      32, SS - 1);
      v8f p2a = {}, p2b = {};
      p2a = wmma_bf16(aq0, ea0, p2a);
      p2a = wmma_bf16(aq1, ea1, p2a);
      p2b = wmma_bf16(aq0, eb0, p2b);
      p2b = wmma_bf16(aq1, eb1, p2b);

#pragma unroll
      for (int r = 0; r < 8; ++r) {
        int Mr  = r + 8 * hi;
        int src = (((hi << 4) | ((Mr - Nl) & 15)) << 2);
        float ga = __int_as_float(
            __builtin_amdgcn_ds_bpermute(src, __float_as_int(p2a[r])));
        float gb = __int_as_float(
            __builtin_amdgcn_ds_bpermute(src, __float_as_int(p2b[r])));
        int gi = iw0 + Mr, gj = jt + Nl;
        float skew = (gi >= gj) ? ((Nl <= Mr) ? gb : ga) : 0.f;
        s[r] += skew;
      }
      stile[tj] = s;
    }

    // ---- online softmax: rows live in (reg, 16-lane half) ----
    float mnew[8];
#pragma unroll
    for (int r = 0; r < 8; ++r) {
      float mx = m8[r];
#pragma unroll
      for (int tj = 0; tj < 4; ++tj) mx = fmaxf(mx, stile[tj][r]);
#pragma unroll
      for (int off = 1; off < 16; off <<= 1)
        mx = fmaxf(mx, __shfl_xor(mx, off, 32));
      mnew[r] = mx;
    }
#pragma unroll
    for (int r = 0; r < 8; ++r) {
      float sc = __expf(m8[r] - mnew[r]);
      l8[r] *= sc;
#pragma unroll
      for (int d = 0; d < 4; ++d) o[d][r] *= sc;
      m8[r] = mnew[r];
    }
    float ls[8] = {0.f, 0.f, 0.f, 0.f, 0.f, 0.f, 0.f, 0.f};
#pragma unroll
    for (int tj = 0; tj < 4; ++tj)
#pragma unroll
      for (int r = 0; r < 8; ++r) {
        float pv = __expf(stile[tj][r] - mnew[r]);
        ls[r] += pv;
        Pt[w][(r + 8 * hi) * 64 + tj * 16 + Nl] = (__bf16)pv;
      }
#pragma unroll
    for (int r = 0; r < 8; ++r) {
      float t = ls[r];
#pragma unroll
      for (int off = 1; off < 16; off <<= 1) t += __shfl_xor(t, off, 32);
      l8[r] += t;
    }

    // ---- O += P~ @ V  (P~ from per-wave LDS, Vt shared) ----
#pragma unroll
    for (int kc = 0; kc < 2; ++kc) {
      v16bf ap = ld_frag(&Pt[w][0], 64, 0, kc * 32);
#pragma unroll
      for (int d = 0; d < 4; ++d) {
        v16bf bvf = ld_frag(&Vt[0], 72, d * 16, kc * 32);
        o[d] = wmma_bf16(ap, bvf, o[d]);
      }
    }
    __syncthreads();  // protect Kl/Vt before next iteration's overwrite
  }

  // ---- normalize and write bf16 output in (B,S,H*DH) layout ----
#pragma unroll
  for (int d = 0; d < 4; ++d)
#pragma unroll
    for (int r = 0; r < 8; ++r) {
      int gi  = iw0 + r + 8 * hi;
      int col = h * DHH + d * 16 + Nl;
      attnb[(size_t)(b * SS + gi) * DD + col] = (__bf16)(o[d][r] / l8[r]);
    }
}

// ---------------------------------------------------------------------------
// Host-side launch sequence
// ---------------------------------------------------------------------------
extern "C" void kernel_launch(void* const* d_in, const int* in_sizes, int n_in,
                              void* d_out, int out_size, void* d_ws, size_t ws_size,
                              hipStream_t stream) {
  (void)in_sizes; (void)n_in; (void)out_size; (void)ws_size;
  const float* x   = (const float*)d_in[0];
  const float* Wq  = (const float*)d_in[1];
  const float* bq  = (const float*)d_in[2];
  const float* Wk  = (const float*)d_in[3];
  const float* bk  = (const float*)d_in[4];
  const float* Wv  = (const float*)d_in[5];
  const float* bv  = (const float*)d_in[6];
  const float* Wo  = (const float*)d_in[7];
  const float* bo  = (const float*)d_in[8];
  const float* Er  = (const float*)d_in[9];
  const float* W1  = (const float*)d_in[10];
  const float* b1  = (const float*)d_in[11];
  const float* W2  = (const float*)d_in[12];
  const float* b2  = (const float*)d_in[13];
  const float* g1  = (const float*)d_in[14];
  const float* be1 = (const float*)d_in[15];
  const float* g2  = (const float*)d_in[16];
  const float* be2 = (const float*)d_in[17];

  char* p = (char*)d_ws;
  auto carve = [&](size_t bytes) -> char* {
    char* r = p;
    p += (bytes + 255) & ~(size_t)255;
    return r;
  };
  float*  x1f  = (float*) carve((size_t)TOK * DD * 4);
  float*  x2f  = (float*) carve((size_t)TOK * DD * 4);
  float*  x3f  = (float*) carve((size_t)TOK * DD * 4);
  __bf16* x1b  = (__bf16*)carve((size_t)TOK * DD * 2);
  __bf16* x3b  = (__bf16*)carve((size_t)TOK * DD * 2);
  __bf16* Wqt  = (__bf16*)carve((size_t)DD * DD * 2);
  __bf16* Wkt  = (__bf16*)carve((size_t)DD * DD * 2);
  __bf16* Wvt  = (__bf16*)carve((size_t)DD * DD * 2);
  __bf16* Wot  = (__bf16*)carve((size_t)DD * DD * 2);
  __bf16* W1t  = (__bf16*)carve((size_t)DD * FFD * 2);
  __bf16* W2t  = (__bf16*)carve((size_t)FFD * DD * 2);
  __bf16* Esb  = (__bf16*)carve((size_t)SS * DHH * 2);
  __bf16* qbuf = (__bf16*)carve((size_t)TOK * DD * 2);
  __bf16* kbuf = (__bf16*)carve((size_t)TOK * DD * 2);
  __bf16* vbuf = (__bf16*)carve((size_t)TOK * DD * 2);
  __bf16* atnb = (__bf16*)carve((size_t)TOK * DD * 2);
  __bf16* h1b  = (__bf16*)carve((size_t)TOK * FFD * 2);

  // one-time-per-call weight transposes + bf16 conversion
  transpose_cvt<<<(DD * DD) / 256, 256, 0, stream>>>(Wq, Wqt, DD, DD);
  transpose_cvt<<<(DD * DD) / 256, 256, 0, stream>>>(Wk, Wkt, DD, DD);
  transpose_cvt<<<(DD * DD) / 256, 256, 0, stream>>>(Wv, Wvt, DD, DD);
  transpose_cvt<<<(DD * DD) / 256, 256, 0, stream>>>(Wo, Wot, DD, DD);
  transpose_cvt<<<(DD * FFD) / 256, 256, 0, stream>>>(W1, W1t, DD, FFD);
  transpose_cvt<<<(FFD * DD) / 256, 256, 0, stream>>>(W2, W2t, FFD, DD);
  build_es<<<(SS * DHH) / 256, 256, 0, stream>>>(Er, Esb);

  // x1 = LN(x)
  layernorm_kernel<<<TOK, 256, 0, stream>>>(x, g1, be1, x1f, x1b);

  // q,k,v projections (q scaled by 1/sqrt(DH)=0.125, folded into epilogue)
  gemm_bf16_kernel<<<dim3(DD / 128, TOK / 64), 128, 0, stream>>>(
      x1b, Wqt, bq, nullptr, nullptr, qbuf, TOK, DD, DD, 0.125f, 0);
  gemm_bf16_kernel<<<dim3(DD / 128, TOK / 64), 128, 0, stream>>>(
      x1b, Wkt, bk, nullptr, nullptr, kbuf, TOK, DD, DD, 1.0f, 0);
  gemm_bf16_kernel<<<dim3(DD / 128, TOK / 64), 128, 0, stream>>>(
      x1b, Wvt, bv, nullptr, nullptr, vbuf, TOK, DD, DD, 1.0f, 0);

  // attention with relative-position skew
  flash_attn_kernel<<<dim3(SS / 64, HH, BB), 128, 0, stream>>>(
      qbuf, kbuf, vbuf, Esb, atnb);

  // x2 = x1 + attn @ Wo + bo
  gemm_bf16_kernel<<<dim3(DD / 128, TOK / 64), 128, 0, stream>>>(
      atnb, Wot, bo, x1f, x2f, nullptr, TOK, DD, DD, 1.0f, 0);

  // x3 = LN(x2)
  layernorm_kernel<<<TOK, 256, 0, stream>>>(x2f, g2, be2, x3f, x3b);

  // h1 = relu(x3 @ W1 + b1)
  gemm_bf16_kernel<<<dim3(FFD / 128, TOK / 64), 128, 0, stream>>>(
      x3b, W1t, b1, nullptr, nullptr, h1b, TOK, FFD, DD, 1.0f, 1);

  // y = x3 + h1 @ W2 + b2
  gemm_bf16_kernel<<<dim3(DD / 128, TOK / 64), 128, 0, stream>>>(
      h1b, W2t, b2, x3f, (float*)d_out, nullptr, TOK, DD, FFD, 1.0f, 0);
}
